// ReConv_13529146983048
// MI455X (gfx1250) — matrix operand order repaired and checked by reading, not compiled, not run on hardware
//
#include <hip/hip_runtime.h>
#include <stdint.h>

// ---------------- problem constants (match reference) ----------------
#define ICN   128
#define OCN   256
#define RC    64          // image rows == cols == 64
#define BATCH 16
#define TAPS  576
#define HOUT  62
#define WOUT  62

// ---------------- LDS tile geometry ----------------
#define LCOLS  66         // 64 image cols + 2 halo cols (zero)
#define IPITCH 136        // image tile ic pitch (bf16): 272B rows, 16B aligned
#define WPITCH 136        // weight tile ic pitch (bf16): 272B rows, 16B aligned

typedef __attribute__((ext_vector_type(16))) __bf16 v16bf;
typedef __attribute__((ext_vector_type(8)))  float  v8f;

__device__ __forceinline__ unsigned short f32_bf16(float f) {
  union { float f; unsigned u; } x; x.f = f;
  unsigned r = x.u + 0x7FFFu + ((x.u >> 16) & 1u);   // round-to-nearest-even
  return (unsigned short)(r >> 16);
}

// ---- CDNA5 async copy to LDS (ASYNCcnt engine), per ISA 08_async_tensor ----
__device__ __forceinline__ void async_b128(unsigned ldsOff, const void* g) {
  asm volatile("global_load_async_to_lds_b128 %0, %1, off"
               :: "v"(ldsOff), "v"(g) : "memory");
}
__device__ __forceinline__ void wait_async0() {
  asm volatile("s_wait_asynccnt 0x0" ::: "memory");
}
__device__ __forceinline__ unsigned lds_off(const void* p) {
  // low 32 bits of the generic (flat) LDS address = wave-relative LDS offset
  return (unsigned)(size_t)p;
}

// ---------------- prologue kernels ----------------
__global__ void k_convert(const float* __restrict__ img,
                          unsigned short* __restrict__ dst, int n) {
  int i = blockIdx.x * blockDim.x + threadIdx.x;
  if (i < n) dst[i] = f32_bf16(img[i]);
}

__global__ void k_zero(unsigned short* __restrict__ wd,
                       float* __restrict__ bias, int nwd) {
  int i = blockIdx.x * blockDim.x + threadIdx.x;
  if (i < nwd) wd[i] = 0;
  if (i < OCN) bias[i] = 0.0f;
}

__global__ void k_scatter(const float* __restrict__ wval,
                          const int*   __restrict__ widx,
                          const int*   __restrict__ bidx,
                          const float* __restrict__ bval,
                          unsigned short* __restrict__ wd,
                          float* __restrict__ bias) {
  int i = blockIdx.x * blockDim.x + threadIdx.x;
  if (i < OCN * TAPS) {
    int oc  = i / TAPS;
    int idx = widx[i];                 // c*(64*64) + ky*64 + kx
    int c   = idx >> 12;
    int rem = idx & 4095;
    int ky  = rem >> 6;
    int kx  = rem & 63;                // < 3 by construction
    // dense layout: [oc][tap = ky*3+kx][ic]
    wd[((size_t)oc * 9 + (ky * 3 + kx)) * ICN + c] = f32_bf16(wval[i]);
  }
  if (i < 128) bias[bidx[i]] = bval[i];  // indices distinct; rest stay zero
}

// ---------------- main WMMA implicit-GEMM conv ----------------
// grid: (2 oc-tiles, 62 rows, 16 batches), block: 256 threads = 8 waves.
// Images: staged once per block, transposed [row][col][ic] in LDS.
// Weights: per-tap 32KB slabs double-buffered in LDS via async-to-LDS DMA.
__global__ __launch_bounds__(256)
void k_conv_wmma(const unsigned short* __restrict__ imgbf,
                 const unsigned short* __restrict__ wd,
                 const float* __restrict__ bias,
                 float* __restrict__ out) {
  extern __shared__ __align__(16) unsigned short smem[];
  unsigned short* ldsImg = smem;                              // 3*66*136 bf16
  unsigned short* ldsW   = smem + 3 * LCOLS * IPITCH;         // 2 * 128*136 bf16

  const int tid  = threadIdx.x;
  const int lane = tid & 31;
  const int wv   = tid >> 5;            // 0..7
  const int ocTile = blockIdx.x;        // 0..1
  const int y    = blockIdx.y;          // output row 0..61
  const int b    = blockIdx.z;          // batch

  // ---- stage 3 input rows, b128 global reads, transpose to [row][col][ic] ----
  for (int e = tid; e < 3 * ICN * 8; e += 256) {      // (r, ic, 8-col segment)
    int r   = e >> 10;                                // / (128*8)
    int rem = e & 1023;
    int ic  = rem >> 3;
    int seg = rem & 7;
    const uint4 q = *(const uint4*)(imgbf +
        (((size_t)b * ICN + ic) * RC + (y + r)) * RC + seg * 8);
    const unsigned short* s = (const unsigned short*)&q;
#pragma unroll
    for (int c = 0; c < 8; ++c)
      ldsImg[(r * LCOLS + seg * 8 + c) * IPITCH + ic] = s[c];
  }
  // zero the 2 halo columns
  for (int e = tid; e < 3 * 2 * ICN; e += 256) {
    int ic = e & 127, ch = (e >> 7) & 1, r = e >> 8;
    ldsImg[(r * LCOLS + 64 + ch) * IPITCH + ic] = 0;
  }

  // ---- async-stage tap 0 weight slab (128 oc x 128 ic bf16) into buffer 0 ----
  {
    const unsigned base = lds_off(ldsW);
    for (int e = tid; e < 2048; e += 256) {           // 128 rows * 16 b128 segs
      int row = e >> 4, seg = e & 15;
      async_b128(base + (unsigned)(row * WPITCH + seg * 8) * 2,
                 wd + ((size_t)(ocTile * 128 + row) * 9 + 0) * ICN + seg * 8);
    }
  }
  wait_async0();
  __syncthreads();

  const int ocBase = ocTile * 128 + wv * 16;          // wave's first oc row
  const int m  = lane & 15;
  const int hi = lane >> 4;                           // half-wave select
  const int kA = hi * 8;                              // A-frag K base (16b layout)

  v8f acc[4] = {v8f{}, v8f{}, v8f{}, v8f{}};

  for (int t = 0; t < 9; ++t) {
    // issue async copy of next tap's weights into the other buffer
    if (t < 8) {
      const unsigned nb = lds_off(ldsW + ((t + 1) & 1) * (ICN * WPITCH));
      for (int e = tid; e < 2048; e += 256) {
        int row = e >> 4, seg = e & 15;
        async_b128(nb + (unsigned)(row * WPITCH + seg * 8) * 2,
                   wd + ((size_t)(ocTile * 128 + row) * 9 + (t + 1)) * ICN + seg * 8);
      }
    }

    const int ky = t / 3, kx = t - ky * 3;
    const unsigned short* wrow =
        ldsW + (t & 1) * (ICN * WPITCH) + (wv * 16 + m) * WPITCH;
#pragma unroll
    for (int icc = 0; icc < 4; ++icc) {
      const int ic0 = icc * 32;
      // A fragment: 16x32 bf16 from LDS; K chunks {kA..kA+7, kA+16..kA+23}
      union { uint4 q[2]; v16bf v; } A;
      A.q[0] = *(const uint4*)(wrow + ic0 + kA);
      A.q[1] = *(const uint4*)(wrow + ic0 + kA + 16);

      const int kB = ic0 + hi * 8;                    // B-frag: ic chunks {kB..kB+7, kB+16..kB+23}
      const unsigned short* lrow = ldsImg + (ky * LCOLS) * IPITCH + kB;
#pragma unroll
      for (int nt = 0; nt < 4; ++nt) {
        const unsigned short* p = lrow + (nt * 16 + m + kx) * IPITCH;
        union { uint4 q[2]; v16bf v; } Bf;
        Bf.q[0] = *(const uint4*)(p);
        Bf.q[1] = *(const uint4*)(p + 16);
        acc[nt] = __builtin_amdgcn_wmma_f32_16x16x32_bf16(
            false, A.v, false, Bf.v, (short)0, acc[nt], false, false);
      }
    }

    wait_async0();       // next-tap slab fully in LDS (this wave's issues)
    __syncthreads();     // all waves done writing + done reading old slab
  }

  // ---- epilogue: C/D layout -> NCHW + bias, mask x >= 62 ----
  float bv[8];
#pragma unroll
  for (int j = 0; j < 8; ++j) bv[j] = bias[ocBase + hi * 8 + j];

#pragma unroll
  for (int nt = 0; nt < 4; ++nt) {
    const int x = nt * 16 + m;
    if (x < WOUT) {
      float* op = out + ((((size_t)b * OCN + ocBase + hi * 8) * HOUT) + y) * WOUT + x;
#pragma unroll
      for (int j = 0; j < 8; ++j)
        op[(size_t)j * HOUT * WOUT] = acc[nt][j] + bv[j];
    }
  }
}

// ---------------- launcher ----------------
extern "C" void kernel_launch(void* const* d_in, const int* in_sizes, int n_in,
                              void* d_out, int out_size, void* d_ws, size_t ws_size,
                              hipStream_t stream) {
  const float* images = (const float*)d_in[0];
  const float* wval   = (const float*)d_in[1];
  const int*   widx   = (const int*)d_in[2];
  // d_in[3] = filter_lengths (uniform 576), d_in[4] = start_points: unused
  const int*   bidx   = (const int*)d_in[5];
  const float* bval   = (const float*)d_in[6];

  const size_t NIMG = (size_t)BATCH * ICN * RC * RC;   // 8388608
  const int    NWD  = OCN * 9 * ICN;                   // 294912

  unsigned short* imgbf = (unsigned short*)d_ws;       // 16.8 MB bf16 images
  unsigned short* wdn   = imgbf + NIMG;                // dense bf16 weights
  float*          biasd = (float*)(wdn + NWD);         // dense f32 bias

  k_convert<<<(int)((NIMG + 255) / 256), 256, 0, stream>>>(images, imgbf, (int)NIMG);
  k_zero<<<(NWD + 255) / 256, 256, 0, stream>>>(wdn, biasd, NWD);
  k_scatter<<<(OCN * TAPS + 255) / 256, 256, 0, stream>>>(wval, widx, bidx, bval, wdn, biasd);

  const size_t smemBytes =
      (size_t)(3 * LCOLS * IPITCH + 2 * ICN * WPITCH) * sizeof(unsigned short); // 123488
  dim3 grid(2, HOUT, BATCH);
  k_conv_wmma<<<grid, 256, smemBytes, stream>>>(imgbf, wdn, biasd, (float*)d_out);
}
